// MultiHeadAttention_24240795419071
// MI455X (gfx1250) — compile-verified, matrix operand order checked by reading, not run on hardware
//
#include <hip/hip_runtime.h>
#include <hip/hip_bf16.h>
#include <math.h>

// ---------------------------------------------------------------------------
// Low-rank MHA on MI455X (gfx1250, wave32, WMMA).
// Reference: Q=q@Wq+b [B,S,16]; K=k@Wk+b; V=k@Wv+b [B,S,2048];
//            P=softmax(Q K^T / sqrt(128)); out = (P V) @ Wo + b.
// Heavy GEMMs: v_wmma_f32_16x16x32_f16 (f16 in, f32 accumulate), with the
// shared B slab staged into LDS via global_load_async_to_lds_b128 (ASYNCcnt),
// double-buffered across the K loop.
// ---------------------------------------------------------------------------

#define HIDDEN 2048
#define HEADS  16
#define NB     4
#define SEQ    2048
#define NROW   (NB * SEQ)   // 8192 flattened rows
#define QLD    32           // Qh/Kh row stride (heads padded 16 -> 32 with zeros)

typedef __attribute__((ext_vector_type(16))) _Float16 v16h;
typedef __attribute__((ext_vector_type(8)))  _Float16 v8h;
typedef __attribute__((ext_vector_type(8)))  float    v8f;

__device__ __forceinline__ v8f wmma16(v16h a, v16h b, v8f c) {
  // D = A(16x32 f16) * B(32x16 f16) + C(16x16 f32)
  return __builtin_amdgcn_wmma_f32_16x16x32_f16(false, a, false, b, (short)0, c,
                                                false, false);
}

// Async copy of 16B/lane from global to LDS (GLOBAL_LOAD_ASYNC_TO_LDS_B128,
// tracked with ASYNCcnt). lds_addr is the wave-relative LDS byte address
// (low 32 bits of the flat shared pointer per the LDS aperture mapping).
__device__ __forceinline__ void async_b128_to_lds(unsigned lds_addr,
                                                  const void* gaddr) {
  asm volatile("global_load_async_to_lds_b128 %0, %1, off"
               :: "v"(lds_addr), "v"(gaddr)
               : "memory");
}
#define WAIT_ASYNC(n) asm volatile("s_wait_asynccnt " #n ::: "memory")

// WMMA 16-bit A/B operand load from row-major f16 storage where the 16-wide
// dim (M for A, N for B) indexes rows and K is contiguous within a row.
// ISA layout: lanes 0-15 hold K={0..7,16..23}, lanes 16-31 hold K={8..15,24..31}.
__device__ __forceinline__ v16h load_op_f16(const _Float16* __restrict__ src,
                                            long row_base, long ld, int k0) {
  int lane = threadIdx.x & 31;
  long row = row_base + (lane & 15);
  int  kA  = k0 + ((lane & 16) ? 8 : 0);
  const _Float16* p = src + row * ld + kA;
  v8h lo = *(const v8h*)p;
  v8h hi = *(const v8h*)(p + 16);
  v16h r;
#pragma unroll
  for (int i = 0; i < 8; ++i) { r[i] = lo[i]; r[i + 8] = hi[i]; }
  return r;
}

// Same operand layout, but source is f32 (converted to f16 on the fly).
__device__ __forceinline__ v16h load_op_f32(const float* __restrict__ src,
                                            long row_base, long ld, int k0) {
  int lane = threadIdx.x & 31;
  long row = row_base + (lane & 15);
  int  kA  = k0 + ((lane & 16) ? 8 : 0);
  const float* p = src + row * ld + kA;
  v16h r;
#pragma unroll
  for (int i = 0; i < 8; ++i) r[i] = (_Float16)p[i];
#pragma unroll
  for (int i = 0; i < 8; ++i) r[i + 8] = (_Float16)p[16 + i];
  return r;
}

// Stage a 64(N) x 32(K) f16 B slab of WT (row-major [N][K], ld=HIDDEN) into
// LDS [64][32]. 128 threads x 2 x 16B = 4KB; per wave: 2 async instructions.
__device__ __forceinline__ void stage_btile(const _Float16* __restrict__ WT,
                                            int n0, int k0,
                                            _Float16* tile /*LDS*/) {
  int t = threadIdx.x;  // 0..127
#pragma unroll
  for (int h = 0; h < 2; ++h) {
    int s = t + h * 128;   // 0..255 segments of 16B
    int row = s >> 2;      // 0..63
    int c16 = s & 3;       // 16B chunk within the 64B row
    const _Float16* g = WT + (long)(n0 + row) * HIDDEN + k0 + c16 * 8;
    unsigned l = (unsigned)(uintptr_t)(tile + row * 32 + c16 * 8);
    async_b128_to_lds(l, g);
  }
}

// ---------------------------------------------------------------------------
// K1: transpose-convert weights: in f32 [K][N] row-major -> out f16 [N][K].
// One-time ~34 MB traffic; L2 (192 MB) absorbs the strided reads.
// ---------------------------------------------------------------------------
__global__ void transpose_f32_to_f16(const float* __restrict__ in,
                                     _Float16* __restrict__ out,
                                     int K, int N) {
  long total = (long)K * N;
  for (long idx = (long)blockIdx.x * blockDim.x + threadIdx.x; idx < total;
       idx += (long)gridDim.x * blockDim.x) {
    long n = idx / K, k = idx - n * K;
    out[idx] = (_Float16)in[k * (long)N + n];
  }
}

// ---------------------------------------------------------------------------
// K2: Q/K head projections. One wave per 16-row tile, N=16 (one WMMA column).
// Qh gets bias and the 1/sqrt(128) logit scale folded in; heads 16..31 = 0.
// ---------------------------------------------------------------------------
__global__ void __launch_bounds__(32)
proj_qk_kernel(const float* __restrict__ q, const float* __restrict__ k,
               const _Float16* __restrict__ WqT, const _Float16* __restrict__ WkT,
               const float* __restrict__ bq, const float* __restrict__ bk,
               _Float16* __restrict__ Qh, _Float16* __restrict__ Kh) {
  const bool isK = (blockIdx.y != 0);
  const float*    x    = isK ? k   : q;
  const _Float16* Wt   = isK ? WkT : WqT;
  const float*    bias = isK ? bk  : bq;
  _Float16*       dst  = isK ? Kh  : Qh;
  const float scale = isK ? 1.0f : 0.08838834764831844f;  // 1/sqrt(128)

  long m0 = (long)blockIdx.x * 16;
  v8f acc = {};
  for (int k0 = 0; k0 < HIDDEN; k0 += 32) {
    v16h a = load_op_f32(x, m0, HIDDEN, k0);
    v16h b = load_op_f16(Wt, 0, HIDDEN, k0);
    acc = wmma16(a, b, acc);
  }
  int lane = threadIdx.x & 31;
  int col = lane & 15;
  int rowoff = (lane & 16) ? 8 : 0;
  float bb = bias[col];
#pragma unroll
  for (int r = 0; r < 8; ++r) {
    long row = m0 + r + rowoff;
    dst[row * QLD + col]      = (_Float16)((acc[r] + bb) * scale);
    dst[row * QLD + 16 + col] = (_Float16)0.0f;  // K-padding
  }
}

// ---------------------------------------------------------------------------
// K3: V projection (V = k @ Wv + b), stored TRANSPOSED as VhT[b][d][s] f16 so
// the attention pass reads its B operand contiguously along K=s.
// Block = 4 waves, 64x64 tile. B slab async-staged to LDS (double buffered);
// epilogue: LDS transpose then coalesced 16B stores.
// ---------------------------------------------------------------------------
__global__ void __launch_bounds__(128)
proj_v_kernel(const float* __restrict__ kin, const _Float16* __restrict__ WvT,
              const float* __restrict__ bias, _Float16* __restrict__ VhT) {
  __shared__ __align__(16) _Float16 Bt[2][64 * 32];  // double-buffered B slab
  __shared__ __align__(16) _Float16 ldsT[64 * 64];   // [n][m] transpose buffer
  int wave = threadIdx.x >> 5;
  long m0 = (long)blockIdx.y * 64 + wave * 16;       // flattened row b*S+s
  int n0 = blockIdx.x * 64;                          // hidden dim d
  v8f acc[4] = {};

  stage_btile(WvT, n0, 0, Bt[0]);
  for (int k0 = 0; k0 < HIDDEN; k0 += 32) {
    int cur = (k0 >> 5) & 1;
    v16h a = load_op_f32(kin, m0, HIDDEN, k0);  // overlap with async wait
    if (k0 + 32 < HIDDEN) {
      stage_btile(WvT, n0, k0 + 32, Bt[cur ^ 1]);
      WAIT_ASYNC(2);   // retire current buffer; next buffer stays in flight
    } else {
      WAIT_ASYNC(0);
    }
    __syncthreads();   // publish all waves' staged portions
#pragma unroll
    for (int j = 0; j < 4; ++j) {
      v16h b = load_op_f16(Bt[cur], 16 * j, 32, 0);  // ds_load_b128 x2
      acc[j] = wmma16(a, b, acc[j]);
    }
    __syncthreads();   // all reads of Bt[cur] done before it is re-staged
  }

  int lane = threadIdx.x & 31;
  int col = lane & 15;
  int rowoff = (lane & 16) ? 8 : 0;
#pragma unroll
  for (int j = 0; j < 4; ++j) {
    float bb = bias[n0 + 16 * j + col];
#pragma unroll
    for (int r = 0; r < 8; ++r) {
      int mloc = wave * 16 + r + rowoff;
      int nloc = 16 * j + col;
      ldsT[nloc * 64 + mloc] = (_Float16)(acc[j][r] + bb);
    }
  }
  __syncthreads();
  long bidx = ((long)blockIdx.y * 64) / SEQ;
  long s0   = ((long)blockIdx.y * 64) % SEQ;
  int t = threadIdx.x;
  int nloc = t >> 1;       // 64 d-rows, 2 threads each
  int half = t & 1;        // each thread moves 32 halves (64B)
  const _Float16* srcp = &ldsT[nloc * 64 + half * 32];
  _Float16* dstp = VhT + (bidx * HIDDEN + n0 + nloc) * (long)SEQ + s0 + half * 32;
#pragma unroll
  for (int i = 0; i < 4; ++i)
    *(v8h*)(dstp + i * 8) = *(const v8h*)(srcp + i * 8);
}

// ---------------------------------------------------------------------------
// K4: per-row softmax stats (running max m, running sum-exp l) over the full
// score row. Scores via WMMA (K=16 heads zero-padded to 32); per-row reduction
// via 16-lane butterfly shuffles (rows live in lane-halves per C/D layout).
// ---------------------------------------------------------------------------
__global__ void __launch_bounds__(32)
softmax_stats_kernel(const _Float16* __restrict__ Qh,
                     const _Float16* __restrict__ Kh,
                     float* __restrict__ mrow, float* __restrict__ lrow) {
  long g0 = (long)blockIdx.x * 16;
  long kbase = (g0 / SEQ) * SEQ;
  int lane = threadIdx.x & 31;
  v16h aq = load_op_f16(Qh, g0, QLD, 0);
  float m_run[8], l_run[8];
#pragma unroll
  for (int r = 0; r < 8; ++r) { m_run[r] = -1e30f; l_run[r] = 0.0f; }
  for (int j = 0; j < SEQ; j += 16) {
    v16h bk = load_op_f16(Kh, kbase + j, QLD, 0);
    v8f z = {};
    v8f s = wmma16(aq, bk, z);
#pragma unroll
    for (int r = 0; r < 8; ++r) {
      float tmax = s[r];
#pragma unroll
      for (int d = 1; d < 16; d <<= 1)
        tmax = fmaxf(tmax, __shfl_xor(tmax, d, 32));
      float mnew = fmaxf(m_run[r], tmax);
      float p = __expf(s[r] - mnew);
#pragma unroll
      for (int d = 1; d < 16; d <<= 1)
        p += __shfl_xor(p, d, 32);
      l_run[r] = l_run[r] * __expf(m_run[r] - mnew) + p;
      m_run[r] = mnew;
    }
  }
  if ((lane & 15) == 0) {
    int rowoff = (lane & 16) ? 8 : 0;
#pragma unroll
    for (int r = 0; r < 8; ++r) {
      mrow[g0 + r + rowoff] = m_run[r];
      lrow[g0 + r + rowoff] = l_run[r];
    }
  }
}

// ---------------------------------------------------------------------------
// K5: attention second pass. For each (16 q-rows) x (64 hidden cols) tile:
// recompute score tiles, apply exp(s-m)/l, bounce P through LDS to convert
// the C/D layout into the A-operand layout, then WMMA against VhT (K over S).
// ---------------------------------------------------------------------------
__global__ void __launch_bounds__(32)
attn_pv_kernel(const _Float16* __restrict__ Qh, const _Float16* __restrict__ Kh,
               const _Float16* __restrict__ VhT, const float* __restrict__ mrow,
               const float* __restrict__ lrow, _Float16* __restrict__ vals) {
  __shared__ __align__(16) _Float16 Pl[16 * 32];  // P tile, rows x K=32
  int d0 = blockIdx.x * 64;
  long b  = blockIdx.z;
  long gq = b * SEQ + (long)blockIdx.y * 16;
  long kbase = b * SEQ;
  int lane = threadIdx.x & 31;
  int col = lane & 15;
  int rowoff = (lane & 16) ? 8 : 0;

  float mr[8], li[8];
#pragma unroll
  for (int r = 0; r < 8; ++r) {
    mr[r] = mrow[gq + r + rowoff];
    li[r] = 1.0f / lrow[gq + r + rowoff];
  }
  v16h aq = load_op_f16(Qh, gq, QLD, 0);
  const _Float16* Vb = VhT + b * (long)HIDDEN * SEQ;
  v8f acc[4] = {};

  for (int k0 = 0; k0 < SEQ; k0 += 32) {
#pragma unroll
    for (int jj = 0; jj < 2; ++jj) {
      v16h bk = load_op_f16(Kh, kbase + k0 + jj * 16, QLD, 0);
      v8f z = {};
      v8f sc = wmma16(aq, bk, z);
#pragma unroll
      for (int r = 0; r < 8; ++r) {
        float p = __expf(sc[r] - mr[r]) * li[r];
        Pl[(r + rowoff) * 32 + jj * 16 + col] = (_Float16)p;
      }
    }
    __syncthreads();  // single-wave group: waitcnt + nop
    v16h pa;
    {
      const _Float16* p = &Pl[(lane & 15) * 32 + ((lane & 16) ? 8 : 0)];
      v8h lo = *(const v8h*)p;
      v8h hi = *(const v8h*)(p + 16);
#pragma unroll
      for (int i = 0; i < 8; ++i) { pa[i] = lo[i]; pa[i + 8] = hi[i]; }
    }
#pragma unroll
    for (int j = 0; j < 4; ++j) {
      v16h bv = load_op_f16(Vb, d0 + 16 * j, SEQ, k0);
      acc[j] = wmma16(pa, bv, acc[j]);
    }
    __syncthreads();  // WAR guard on Pl before next iteration's stores
  }
#pragma unroll
  for (int j = 0; j < 4; ++j)
#pragma unroll
    for (int r = 0; r < 8; ++r)
      vals[(gq + r + rowoff) * (long)HIDDEN + d0 + 16 * j + col] =
          (_Float16)acc[j][r];
}

// ---------------------------------------------------------------------------
// K6: output projection: out = vals(f16) @ WoT + b, f32 result to d_out.
// Same async-staged LDS B pipeline as K3.
// ---------------------------------------------------------------------------
__global__ void __launch_bounds__(128)
out_proj_kernel(const _Float16* __restrict__ vals,
                const _Float16* __restrict__ WoT,
                const float* __restrict__ bias, float* __restrict__ out) {
  __shared__ __align__(16) _Float16 Bt[2][64 * 32];
  int wave = threadIdx.x >> 5;
  long m0 = (long)blockIdx.y * 64 + wave * 16;
  int n0 = blockIdx.x * 64;
  v8f acc[4] = {};

  stage_btile(WoT, n0, 0, Bt[0]);
  for (int k0 = 0; k0 < HIDDEN; k0 += 32) {
    int cur = (k0 >> 5) & 1;
    v16h a = load_op_f16(vals, m0, HIDDEN, k0);
    if (k0 + 32 < HIDDEN) {
      stage_btile(WoT, n0, k0 + 32, Bt[cur ^ 1]);
      WAIT_ASYNC(2);
    } else {
      WAIT_ASYNC(0);
    }
    __syncthreads();
#pragma unroll
    for (int j = 0; j < 4; ++j) {
      v16h b = load_op_f16(Bt[cur], 16 * j, 32, 0);
      acc[j] = wmma16(a, b, acc[j]);
    }
    __syncthreads();
  }

  int lane = threadIdx.x & 31;
  int col = lane & 15;
  int rowoff = (lane & 16) ? 8 : 0;
#pragma unroll
  for (int j = 0; j < 4; ++j) {
    float bb = bias[n0 + 16 * j + col];
#pragma unroll
    for (int r = 0; r < 8; ++r)
      out[(m0 + r + rowoff) * (long)HIDDEN + n0 + 16 * j + col] =
          acc[j][r] + bb;
  }
}

// ---------------------------------------------------------------------------
extern "C" void kernel_launch(void* const* d_in, const int* in_sizes, int n_in,
                              void* d_out, int out_size, void* d_ws,
                              size_t ws_size, hipStream_t stream) {
  const float* q     = (const float*)d_in[0];
  const float* k     = (const float*)d_in[1];
  // d_in[2] (v) is unused by the reference (V is projected from k).
  const float* wqs_w = (const float*)d_in[3];
  const float* wqs_b = (const float*)d_in[4];
  const float* wks_w = (const float*)d_in[5];
  const float* wks_b = (const float*)d_in[6];
  const float* wvs_w = (const float*)d_in[7];
  const float* wvs_b = (const float*)d_in[8];
  const float* wo_w  = (const float*)d_in[9];
  const float* wo_b  = (const float*)d_in[10];
  float* out = (float*)d_out;

  // Workspace carve (~84 MB total), 256B-aligned slices.
  char* ws = (char*)d_ws;
  size_t off = 0;
  auto carve = [&](size_t bytes) -> char* {
    char* p = ws + off;
    off += (bytes + 255) & ~(size_t)255;
    return p;
  };
  _Float16* WqT  = (_Float16*)carve((size_t)HEADS * HIDDEN * 2);
  _Float16* WkT  = (_Float16*)carve((size_t)HEADS * HIDDEN * 2);
  _Float16* WvT  = (_Float16*)carve((size_t)HIDDEN * HIDDEN * 2);
  _Float16* WoT  = (_Float16*)carve((size_t)HIDDEN * HIDDEN * 2);
  _Float16* Qh   = (_Float16*)carve((size_t)NROW * QLD * 2);
  _Float16* Kh   = (_Float16*)carve((size_t)NROW * QLD * 2);
  _Float16* VhT  = (_Float16*)carve((size_t)NROW * HIDDEN * 2);
  _Float16* vals = (_Float16*)carve((size_t)NROW * HIDDEN * 2);
  float*    mrow = (float*)carve((size_t)NROW * 4);
  float*    lrow = (float*)carve((size_t)NROW * 4);
  (void)ws_size; (void)n_in; (void)in_sizes; (void)out_size;

  auto tgrid = [](long total) {
    long b = (total + 255) / 256;
    return (int)(b > 4096 ? 4096 : b);
  };

  // 1) Weight transpose+convert to f16 [N][K].
  transpose_f32_to_f16<<<tgrid((long)HIDDEN * HEADS), 256, 0, stream>>>(
      wqs_w, WqT, HIDDEN, HEADS);
  transpose_f32_to_f16<<<tgrid((long)HIDDEN * HEADS), 256, 0, stream>>>(
      wks_w, WkT, HIDDEN, HEADS);
  transpose_f32_to_f16<<<tgrid((long)HIDDEN * HIDDEN), 256, 0, stream>>>(
      wvs_w, WvT, HIDDEN, HIDDEN);
  transpose_f32_to_f16<<<tgrid((long)HIDDEN * HIDDEN), 256, 0, stream>>>(
      wo_w, WoT, HIDDEN, HIDDEN);

  // 2) Q/K head projections (scale folded into Qh).
  proj_qk_kernel<<<dim3(NROW / 16, 2), 32, 0, stream>>>(
      q, k, WqT, WkT, wqs_b, wks_b, Qh, Kh);

  // 3) V projection, stored transposed.
  proj_v_kernel<<<dim3(HIDDEN / 64, NROW / 64), 128, 0, stream>>>(
      k, WvT, wvs_b, VhT);

  // 4) Softmax row stats.
  softmax_stats_kernel<<<NROW / 16, 32, 0, stream>>>(Qh, Kh, mrow, lrow);

  // 5) P @ V (flash-style second pass).
  attn_pv_kernel<<<dim3(HIDDEN / 64, SEQ / 16, NB), 32, 0, stream>>>(
      Qh, Kh, VhT, mrow, lrow, vals);

  // 6) Output projection.
  out_proj_kernel<<<dim3(HIDDEN / 64, NROW / 64), 128, 0, stream>>>(
      vals, WoT, wo_b, out);
}
